// EnhancedTransformerLayer_84799834293078
// MI455X (gfx1250) — matrix operand order repaired and checked by reading, not compiled, not run on hardware
//
#include <hip/hip_runtime.h>
#include <hip/hip_bf16.h>
#include <cstdint>

// Problem constants (from reference)
#define BB   2
#define SS   2048
#define EE   1024
#define HH   16
#define DD   64
#define NEXP 8
#define MTOK (BB * SS)   // 4096 tokens

typedef __attribute__((ext_vector_type(16))) __bf16 bf16x16;
typedef __attribute__((ext_vector_type(8)))  float  f32x8;

// 32-byte WMMA operand fragment: u16 lanes / two dwordx4 / bf16 vector
union FragU {
    unsigned short u[16];
    uint4          q[2];
    bf16x16        v;
};

__device__ __forceinline__ unsigned short f2bf(float f) {
    unsigned int u = __float_as_uint(f);
    u += 0x7FFFu + ((u >> 16) & 1u);   // round-to-nearest-even
    return (unsigned short)(u >> 16);
}

// ---------------------------------------------------------------------------
// Tensor Data Mover: 2D bf16 tile (tile_w x tile_h) from global -> LDS, with
// LDS row padding so padded row stride matches our aligned fragment layout.
//
// NOTE on the LDS fences below: the TDM intrinsic's LDS writes are invisible
// to alias analysis. If the __shared__ object never escapes, LLVM folds its
// loads to undef (observed rounds 2-3: wmma fed by loop-invariant v[0:7]).
// Therefore (a) the descriptor lds_addr is derived from the *pointer* itself
// (flat shared address bits [31:0] == LDS byte offset, per the aperture
// rules), capturing it, and (b) every fence passes the captured pointer into
// side-effecting asm, so the compiler must assume LDS changed.
// ---------------------------------------------------------------------------
#if defined(__HIP_DEVICE_COMPILE__) && __has_builtin(__builtin_amdgcn_tensor_load_to_lds) && __has_builtin(__builtin_amdgcn_s_wait_tensorcnt)
#define HAVE_TDM 1
#else
#define HAVE_TDM 0
#endif

// compiler-level LDS fence carrying the captured shared-memory pointer
#define SM_FENCE(p) asm volatile("" : "+v"(p) : : "memory")

#if HAVE_TDM
typedef __attribute__((ext_vector_type(4))) unsigned int u32x4;
typedef __attribute__((ext_vector_type(8))) int          i32x8;
typedef __attribute__((ext_vector_type(4))) int          i32x4;

__device__ __forceinline__ void tdm_load_2d_bf16(
    unsigned lds_addr, const unsigned short* gptr,
    unsigned tile_w,      // tile dim0 in elements
    unsigned tile_h,      // tile dim1 in rows
    unsigned row_stride,  // tensor dim0 stride in elements
    unsigned rows_bound,  // tensor dim1 (OOB bound)
    unsigned pad_int_code,// pad after 2^(c+1) DWORDs
    unsigned pad_amt_code)// pad amount (c+1) DWORDs
{
    unsigned long long ga = (unsigned long long)(size_t)gptr;
    u32x4 g0;
    g0[0] = 1u;                                     // count=1 valid descriptor
    g0[1] = lds_addr;                               // LDS byte address
    g0[2] = (unsigned)ga;                           // global addr lo
    g0[3] = (unsigned)((ga >> 32) & 0x01FFFFFFu) | (2u << 30);  // hi | type=2
    i32x8 g1;
    g1[0] = (int)((1u << 16) | (1u << 20) |         // data_size=2B, pad_enable
                  (pad_int_code << 22) | (pad_amt_code << 25));
    g1[1] = (int)((row_stride & 0xFFFFu) << 16);    // tensor_dim0 lo16
    g1[2] = (int)((row_stride >> 16) | ((rows_bound & 0xFFFFu) << 16));
    g1[3] = (int)((rows_bound >> 16) | (tile_w << 16));   // tile_dim0
    g1[4] = (int)(tile_h & 0xFFFFu);                // tile_dim1 (tile_dim2=0)
    g1[5] = (int)row_stride;                        // tensor_dim0_stride lo32
    g1[6] = 0;
    g1[7] = 0;
    i32x4 z = {0, 0, 0, 0};
#if __clang_major__ >= 23
    i32x8 z8 = {0, 0, 0, 0, 0, 0, 0, 0};
    __builtin_amdgcn_tensor_load_to_lds(g0, g1, z, z, z8, 0);
#else
    __builtin_amdgcn_tensor_load_to_lds(g0, g1, z, z, 0);
#endif
}
#endif

// ---------------------------------------------------------------------------
// fp32 -> packed bf16 conversion (whole arrays, done once per launch)
// ---------------------------------------------------------------------------
__global__ __launch_bounds__(256)
void cvt_bf16_kernel(const float* __restrict__ in, unsigned short* __restrict__ out, int n4)
{
    int i = blockIdx.x * 256 + threadIdx.x;
    if (i >= n4) return;
    const float4 f = ((const float4*)in)[i];
    uint2 o;
    o.x = (unsigned)f2bf(f.x) | ((unsigned)f2bf(f.y) << 16);
    o.y = (unsigned)f2bf(f.z) | ((unsigned)f2bf(f.w) << 16);
    ((uint2*)out)[i] = o;
}

// ---------------------------------------------------------------------------
// Generic tiled GEMM on bf16 operands:
//   r = sum_k A[m,k]*W[n,k] + bias[n];  r *= rowscale[m]; r += residual; r += C(old)
// Block = 128 threads (4 waves), 64x64 tile, K-step 32.
// Double-buffered LDS: TDM fetches tile i+1 while WMMA consumes tile i;
// s_wait_tensorcnt(2) retires the oldest (current) descriptor pair.
// ---------------------------------------------------------------------------
__global__ __launch_bounds__(128)
void gemm_wmma_bf16(const unsigned short* __restrict__ Abf,
                    const unsigned short* __restrict__ Wbf,
                    const float* __restrict__ bias,
                    float* __restrict__ C,
                    unsigned short* __restrict__ Cbf,
                    int M, int N, int Kd,
                    const float* __restrict__ rowscale,
                    const float* __restrict__ residual,
                    int accumulate)
{
    // padded row stride 40 u16 (80B): 16B-aligned fragments + bank skew
    __shared__ struct {
        unsigned short As[2][64][40];   // +0,      +5120
        unsigned short Bs[2][64][40];   // +10240,  +15360
    } sm;
    enum { TILE_B = 64 * 40 * 2, BS_OFF = 2 * TILE_B };

    char* smp = (char*)&sm;                          // captured LDS pointer
    const unsigned lds0 = (unsigned)(size_t)smp;     // LDS byte addr of sm

    const int tid  = threadIdx.x;
    const int wave = tid >> 5;
    const int lane = tid & 31;
    const int g    = lane >> 4;   // 16-lane half
    const int hl   = lane & 15;
    const int bm   = blockIdx.y * 64;
    const int bn   = blockIdx.x * 64;

    f32x8 acc[4];
#pragma unroll
    for (int nt = 0; nt < 4; ++nt)
#pragma unroll
        for (int e = 0; e < 8; ++e) acc[nt][e] = 0.f;

#if HAVE_TDM
    if (wave == 0) {   // prologue: fetch first tile pair into buffer 0
        // 32 elems/row -> 16 DWORDs (code 3), pad 4 DWORDs (code 3) => 80B rows
        tdm_load_2d_bf16(lds0,          Abf + (size_t)bm * Kd, 32, 64, Kd, 64, 3, 3);
        tdm_load_2d_bf16(lds0 + BS_OFF, Wbf + (size_t)bn * Kd, 32, 64, Kd, 64, 3, 3);
    }
#endif

    int cur = 0;
    for (int k0 = 0; k0 < Kd; k0 += 32, cur ^= 1) {
        __syncthreads();          // reads of the buffer we are about to fill are done
        SM_FENCE(smp);
#if HAVE_TDM
        if (wave == 0) {
            if (k0 + 32 < Kd) {   // overlap: fetch next pair into other buffer
                int nxt = cur ^ 1;
                tdm_load_2d_bf16(lds0 + (unsigned)(nxt * TILE_B),
                                 Abf + (size_t)bm * Kd + k0 + 32, 32, 64, Kd, 64, 3, 3);
                tdm_load_2d_bf16(lds0 + (unsigned)(BS_OFF + nxt * TILE_B),
                                 Wbf + (size_t)bn * Kd + k0 + 32, 32, 64, Kd, 64, 3, 3);
                __builtin_amdgcn_s_wait_tensorcnt(2);   // current pair retired
            } else {
                __builtin_amdgcn_s_wait_tensorcnt(0);
            }
        }
#else
        for (int idx = tid; idx < 256; idx += 128) {  // 256 x uint4 per buffer
            int r = idx >> 2, c = (idx & 3) * 8;
            *(uint4*)&sm.As[cur][r][c] = *(const uint4*)(Abf + (size_t)(bm + r) * Kd + k0 + c);
            *(uint4*)&sm.Bs[cur][r][c] = *(const uint4*)(Wbf + (size_t)(bn + r) * Kd + k0 + c);
        }
#endif
        __syncthreads();
        SM_FENCE(smp);

        // A fragment: lane -> row (wave*16+hl), K = g*8 + j + (j&8?8:0)
        FragU a;
        {
            const unsigned short* arow = &sm.As[cur][wave * 16 + hl][0];
            a.q[0] = *(const uint4*)(arow + g * 8);
            a.q[1] = *(const uint4*)(arow + g * 8 + 16);
        }
#pragma unroll
        for (int nt = 0; nt < 4; ++nt) {
            // B fragment: lane -> column (nt*16+hl), K = g*8 + j + (j&8?8:0)
            FragU b;
            const unsigned short* brow = &sm.Bs[cur][nt * 16 + hl][0];
            b.q[0] = *(const uint4*)(brow + g * 8);
            b.q[1] = *(const uint4*)(brow + g * 8 + 16);
            acc[nt] = __builtin_amdgcn_wmma_f32_16x16x32_bf16(
                false, a.v, false, b.v, (short)0, acc[nt], false, false);
        }
    }

    // C/D layout: lanes 0-15 -> (M=v, N=hl); lanes 16-31 -> (M=v+8, N=hl)
#pragma unroll
    for (int nt = 0; nt < 4; ++nt)
#pragma unroll
        for (int vv = 0; vv < 8; ++vv) {
            int row = bm + wave * 16 + vv + 8 * g;
            int col = bn + nt * 16 + hl;
            size_t off = (size_t)row * N + col;
            float r = acc[nt][vv] + bias[col];
            if (rowscale)   r *= rowscale[row];
            if (residual)   r += residual[off];
            if (accumulate) r += C[off];
            if (C)   C[off] = r;
            if (Cbf) Cbf[off] = f2bf(r);
        }
}

// ---------------------------------------------------------------------------
// RoPE: read f32 q/k ([B,S,H,D]), write bf16 q/k for the attention kernel.
// ---------------------------------------------------------------------------
__global__ void rope_kernel(const float* __restrict__ qin, const float* __restrict__ kin,
                            unsigned short* __restrict__ qo, unsigned short* __restrict__ ko)
{
    int idx = blockIdx.x * blockDim.x + threadIdx.x;   // B*S*H*32 threads
    int i = idx & 31;
    int h = (idx >> 5) & (HH - 1);
    int s = (idx >> 9) & (SS - 1);
    int b = idx >> 20;
    float inv = __expf(-(float)i * (9.210340371976184f / 32.0f));  // 10000^(-i/32)
    float ang = (float)s * inv;
    float sn, cs;
    __sincosf(ang, &sn, &cs);
    size_t base = ((size_t)(b * SS + s) * HH + h) * DD;
    float q1 = qin[base + i], q2 = qin[base + i + 32];
    qo[base + i]      = f2bf(q1 * cs - q2 * sn);
    qo[base + i + 32] = f2bf(q2 * cs + q1 * sn);
    float k1 = kin[base + i], k2 = kin[base + i + 32];
    ko[base + i]      = f2bf(k1 * cs - k2 * sn);
    ko[base + i + 32] = f2bf(k2 * cs + k1 * sn);
}

// ---------------------------------------------------------------------------
// Flash attention: one wave32 per (b, h, 16-row q tile). K/V tiles streamed
// into double-buffered LDS by the TDM; online softmax; all matmuls on WMMA.
// ---------------------------------------------------------------------------
__global__ __launch_bounds__(32)
void attn_wmma(const unsigned short* __restrict__ q, const unsigned short* __restrict__ k,
               const unsigned short* __restrict__ v,
               float* __restrict__ o, unsigned short* __restrict__ obf)
{
    // Kb/Vb rows: 64 elems + pad 8 -> 144B stride; Pb rows: 32 + pad 8 -> 80B
    __shared__ struct {
        unsigned short Kb[2][32][72];   // +0,    +4608
        unsigned short Vb[2][32][72];   // +9216, +13824
        unsigned short Pb[16][40];      // +18432
    } sm;
    enum { KV_B = 32 * 72 * 2, V_OFF = 2 * KV_B, NKB = SS / 32 };

    char* smp = (char*)&sm;                          // captured LDS pointer
    const unsigned lds0 = (unsigned)(size_t)smp;     // LDS byte addr of sm

    const int lane = threadIdx.x;
    const int g = lane >> 4, hl = lane & 15;
    const int nq = SS / 16;
    const int qt = blockIdx.x % nq;
    const int h  = (blockIdx.x / nq) % HH;
    const int b  = blockIdx.x / (nq * HH);
    const size_t rowstr = (size_t)HH * DD;

    // Q fragments: two contiguous 8-elem runs per dstep -> b128 global loads
    FragU qa[2];
    {
        const unsigned short* qrow = q + ((size_t)(b * SS + qt * 16 + hl) * HH + h) * DD;
#pragma unroll
        for (int dstep = 0; dstep < 2; ++dstep) {
            qa[dstep].q[0] = *(const uint4*)(qrow + dstep * 32 + g * 8);
            qa[dstep].q[1] = *(const uint4*)(qrow + dstep * 32 + g * 8 + 16);
        }
    }

    float m_run[8], l_run[8];
    f32x8 acc[4];
#pragma unroll
    for (int vv = 0; vv < 8; ++vv) { m_run[vv] = -3.0e38f; l_run[vv] = 0.f; }
#pragma unroll
    for (int nt = 0; nt < 4; ++nt)
#pragma unroll
        for (int e = 0; e < 8; ++e) acc[nt][e] = 0.f;

#if HAVE_TDM
    {   // prologue: first K/V tile pair into buffer 0
        const size_t kb0 = ((size_t)(b * SS) * HH + h) * DD;
        // 64 elems/row = 32 DWORDs (code 4), pad 4 DWORDs (code 3) => 144B rows
        tdm_load_2d_bf16(lds0,         k + kb0, 64, 32, HH * DD, 32, 4, 3);
        tdm_load_2d_bf16(lds0 + V_OFF, v + kb0, 64, 32, HH * DD, 32, 4, 3);
    }
#endif

    for (int kb = 0; kb < NKB; ++kb) {
        const int cur = kb & 1;
        SM_FENCE(smp);
#if HAVE_TDM
        if (kb + 1 < NKB) {
            // our ds reads of the buffer being overwritten may still be in
            // flight; TDM is unordered with DS ops -> drain DScnt first
            asm volatile("s_wait_dscnt 0x0" : "+v"(smp) : : "memory");
            const int nxt = cur ^ 1;
            const size_t kbn = ((size_t)(b * SS + (kb + 1) * 32) * HH + h) * DD;
            tdm_load_2d_bf16(lds0 + (unsigned)(nxt * KV_B),         k + kbn, 64, 32, HH * DD, 32, 4, 3);
            tdm_load_2d_bf16(lds0 + (unsigned)(V_OFF + nxt * KV_B), v + kbn, 64, 32, HH * DD, 32, 4, 3);
            __builtin_amdgcn_s_wait_tensorcnt(2);   // current pair retired
        } else {
            __builtin_amdgcn_s_wait_tensorcnt(0);
        }
        SM_FENCE(smp);
#else
        {
            const size_t kbase = ((size_t)(b * SS + kb * 32) * HH + h) * DD;
            const unsigned short* kr = k + kbase + (size_t)lane * rowstr;
            const unsigned short* vr = v + kbase + (size_t)lane * rowstr;
#pragma unroll
            for (int c = 0; c < 8; ++c) {
                *(uint4*)&sm.Kb[cur][lane][c * 8] = *(const uint4*)(kr + c * 8);
                *(uint4*)&sm.Vb[cur][lane][c * 8] = *(const uint4*)(vr + c * 8);
            }
        }
        __syncthreads();
#endif

        // scores S[16x32] = Q(16x64) . K(32x64)^T
        f32x8 sc[2];
#pragma unroll
        for (int sub = 0; sub < 2; ++sub) {
#pragma unroll
            for (int e = 0; e < 8; ++e) sc[sub][e] = 0.f;
#pragma unroll
            for (int dstep = 0; dstep < 2; ++dstep) {
                FragU bk;   // column = key sub*16+hl, K = d
                const unsigned short* rowp = &sm.Kb[cur][sub * 16 + hl][0];
                bk.q[0] = *(const uint4*)(rowp + dstep * 32 + g * 8);
                bk.q[1] = *(const uint4*)(rowp + dstep * 32 + g * 8 + 16);
                sc[sub] = __builtin_amdgcn_wmma_f32_16x16x32_bf16(
                    false, qa[dstep].v, false, bk.v, (short)0, sc[sub], false, false);
            }
        }

        // online softmax (1/sqrt(D) folded into f32 scores here)
#pragma unroll
        for (int vv = 0; vv < 8; ++vv) {
            float s0 = sc[0][vv] * 0.125f;
            float s1 = sc[1][vv] * 0.125f;
            float e = fmaxf(s0, s1);
            e = fmaxf(e, __shfl_xor(e, 1, 32));
            e = fmaxf(e, __shfl_xor(e, 2, 32));
            e = fmaxf(e, __shfl_xor(e, 4, 32));
            e = fmaxf(e, __shfl_xor(e, 8, 32));
            float mn = fmaxf(m_run[vv], e);
            float sf = __expf(m_run[vv] - mn);
            m_run[vv] = mn;
            float p0 = __expf(s0 - mn);
            float p1 = __expf(s1 - mn);
            sc[0][vv] = p0; sc[1][vv] = p1;
            float ps = p0 + p1;
            ps += __shfl_xor(ps, 1, 32);
            ps += __shfl_xor(ps, 2, 32);
            ps += __shfl_xor(ps, 4, 32);
            ps += __shfl_xor(ps, 8, 32);
            l_run[vv] = l_run[vv] * sf + ps;
#pragma unroll
            for (int nt = 0; nt < 4; ++nt) acc[nt][vv] *= sf;
        }

        // relayout P (C layout -> A fragment layout) through LDS
        // (single wave: DS ops are in-order, no barrier needed)
#pragma unroll
        for (int sub = 0; sub < 2; ++sub)
#pragma unroll
            for (int vv = 0; vv < 8; ++vv)
                sm.Pb[vv + 8 * g][sub * 16 + hl] = f2bf(sc[sub][vv]);
        SM_FENCE(smp);

        FragU pa;
        {
            const unsigned short* prow = &sm.Pb[hl][0];
            pa.q[0] = *(const uint4*)(prow + g * 8);
            pa.q[1] = *(const uint4*)(prow + g * 8 + 16);
        }
#pragma unroll
        for (int nt = 0; nt < 4; ++nt) {
            FragU bv;   // column = out dim nt*16+hl, K = key g*8+j+(j&8?8:0)
#pragma unroll
            for (int j = 0; j < 16; ++j)
                bv.u[j] = sm.Vb[cur][g * 8 + (j & 7) + (j >> 3) * 16][nt * 16 + hl];
            acc[nt] = __builtin_amdgcn_wmma_f32_16x16x32_bf16(
                false, pa.v, false, bv.v, (short)0, acc[nt], false, false);
        }
    }

#pragma unroll
    for (int nt = 0; nt < 4; ++nt)
#pragma unroll
        for (int vv = 0; vv < 8; ++vv) {
            int srow = qt * 16 + vv + 8 * g;
            int d    = nt * 16 + hl;
            size_t off = ((size_t)(b * SS + srow) * HH + h) * DD + d;
            float r = acc[nt][vv] / l_run[vv];
            o[off]   = r;
            obf[off] = f2bf(r);
        }
}

// ---------------------------------------------------------------------------
// Gate logits + softmax + top-2 -> per-expert row scales (0 if unselected).
// ---------------------------------------------------------------------------
__global__ __launch_bounds__(256)
void gate_topk(const float* __restrict__ attn, const float* __restrict__ gw,
               const float* __restrict__ gb, float* __restrict__ scale)
{
    __shared__ float logits[NEXP];
    const int t    = blockIdx.x;
    const int e    = threadIdx.x >> 5;
    const int lane = threadIdx.x & 31;
    float dot = 0.f;
    for (int i = lane; i < EE; i += 32)
        dot += attn[(size_t)t * EE + i] * gw[(size_t)e * EE + i];
    for (int m = 16; m; m >>= 1) dot += __shfl_xor(dot, m, 32);
    if (lane == 0) logits[e] = dot + gb[e];
    __syncthreads();
    if (threadIdx.x == 0) {
        float mx = logits[0];
        for (int i = 1; i < NEXP; ++i) mx = fmaxf(mx, logits[i]);
        float p[NEXP], se = 0.f;
        for (int i = 0; i < NEXP; ++i) { p[i] = __expf(logits[i] - mx); se += p[i]; }
        float inv = 1.f / se;
        int i1 = 0;
        for (int i = 1; i < NEXP; ++i) if (p[i] > p[i1]) i1 = i;
        int i2 = (i1 == 0) ? 1 : 0;
        for (int i = 0; i < NEXP; ++i) if (i != i1 && p[i] > p[i2]) i2 = i;
        for (int i = 0; i < NEXP; ++i)
            scale[(size_t)i * MTOK + t] = (i == i1 || i == i2) ? p[i] * inv : 0.f;
    }
}

// ---------------------------------------------------------------------------
extern "C" void kernel_launch(void* const* d_in, const int* in_sizes, int n_in,
                              void* d_out, int out_size, void* d_ws, size_t ws_size,
                              hipStream_t stream)
{
    const float* x        = (const float*)d_in[0];
    const float* q_w      = (const float*)d_in[1];
    const float* q_b      = (const float*)d_in[2];
    const float* k_w      = (const float*)d_in[3];
    const float* k_b      = (const float*)d_in[4];
    const float* v_w      = (const float*)d_in[5];
    const float* v_b      = (const float*)d_in[6];
    const float* gate_w   = (const float*)d_in[7];
    const float* gate_b   = (const float*)d_in[8];
    const float* expert_w = (const float*)d_in[9];
    const float* expert_b = (const float*)d_in[10];
    const float* ffn_w    = (const float*)d_in[11];
    const float* ffn_b    = (const float*)d_in[12];
    float* out = (float*)d_out;

    const size_t ME  = (size_t)MTOK * EE;   // 4M elements (activations)
    const size_t WE  = (size_t)EE * EE;     // 1M elements (one weight matrix)

    // workspace carve-up
    char* p = (char*)d_ws;
    auto alloc_f32 = [&](size_t n) { float* r = (float*)p; p += n * 4; return r; };
    auto alloc_u16 = [&](size_t n) { unsigned short* r = (unsigned short*)p; p += n * 2; return r; };
    float* qbF  = alloc_f32(ME);            // q projection (f32, pre-RoPE)
    float* kbF  = alloc_f32(ME);            // k projection (f32, pre-RoPE)
    float* at   = alloc_f32(ME);            // attention out (f32, for gating)
    float* moe  = alloc_f32(ME);            // MoE accumulator (f32)
    float* scl  = alloc_f32((size_t)NEXP * MTOK);
    unsigned short* x_bf  = alloc_u16(ME);
    unsigned short* qw_bf = alloc_u16(WE);
    unsigned short* kw_bf = alloc_u16(WE);
    unsigned short* vw_bf = alloc_u16(WE);
    unsigned short* ew_bf = alloc_u16((size_t)NEXP * WE);
    unsigned short* fw_bf = alloc_u16(WE);
    unsigned short* q_bf  = alloc_u16(ME);  // post-RoPE q (bf16)
    unsigned short* k_bf  = alloc_u16(ME);  // post-RoPE k (bf16)
    unsigned short* v_bf  = alloc_u16(ME);  // v (bf16)
    unsigned short* at_bf = alloc_u16(ME);  // attention out (bf16)
    unsigned short* mo_bf = alloc_u16(ME);  // MoE out (bf16)

    auto cvt = [&](const float* src, unsigned short* dst, size_t n) {
        int n4 = (int)(n / 4);
        cvt_bf16_kernel<<<(n4 + 255) / 256, 256, 0, stream>>>(src, dst, n4);
    };
    // one-time bf16 conversions of inputs
    cvt(x, x_bf, ME);
    cvt(q_w, qw_bf, WE);
    cvt(k_w, kw_bf, WE);
    cvt(v_w, vw_bf, WE);
    cvt(expert_w, ew_bf, (size_t)NEXP * WE);
    cvt(ffn_w, fw_bf, WE);

    dim3 gg(EE / 64, MTOK / 64);

    // QKV projections (v goes straight to bf16; q,k need f32 for RoPE)
    gemm_wmma_bf16<<<gg, 128, 0, stream>>>(x_bf, qw_bf, q_b, qbF, nullptr,
                                           MTOK, EE, EE, nullptr, nullptr, 0);
    gemm_wmma_bf16<<<gg, 128, 0, stream>>>(x_bf, kw_bf, k_b, kbF, nullptr,
                                           MTOK, EE, EE, nullptr, nullptr, 0);
    gemm_wmma_bf16<<<gg, 128, 0, stream>>>(x_bf, vw_bf, v_b, nullptr, v_bf,
                                           MTOK, EE, EE, nullptr, nullptr, 0);

    // RoPE -> bf16 q,k
    rope_kernel<<<(BB * SS * HH * 32) / 256, 256, 0, stream>>>(qbF, kbF, q_bf, k_bf);

    // attention
    attn_wmma<<<BB * HH * (SS / 16), 32, 0, stream>>>(q_bf, k_bf, v_bf, at, at_bf);

    // gating / top-2 -> per-expert row scales
    gate_topk<<<MTOK, 256, 0, stream>>>(at, gate_w, gate_b, scl);

    // MoE: 8 dense expert GEMMs, row-scaled by gate prob (0 for unselected),
    // accumulated into moe; last launch also emits the bf16 mirror.
    for (int e = 0; e < NEXP; ++e)
        gemm_wmma_bf16<<<gg, 128, 0, stream>>>(at_bf, ew_bf + (size_t)e * WE,
                                               expert_b + (size_t)e * EE,
                                               moe, (e == NEXP - 1) ? mo_bf : nullptr,
                                               MTOK, EE, EE,
                                               scl + (size_t)e * MTOK, nullptr, e ? 1 : 0);

    // final FFN + residual(x)
    gemm_wmma_bf16<<<gg, 128, 0, stream>>>(mo_bf, fw_bf, ffn_b, out, nullptr,
                                           MTOK, EE, EE, nullptr, x, 0);
}
